// Bionic_19696720019998
// MI455X (gfx1250) — compile-verified, matrix operand order checked by reading, not run on hardware
//
#include <hip/hip_runtime.h>
#include <hip/hip_bf16.h>

// ---------------------------------------------------------------------------
// Sizes from the reference
// ---------------------------------------------------------------------------
#define HH 4
#define CC 64
#define DD 256            // H*C
#define N0 32768
#define N1 8192
#define NB 2048
#define MM 3
#define IN_SIZE 50000
#define EMB 512
#define E0 262144
#define E1 65536
#define NEG_SLOPE 0.2f

typedef __attribute__((ext_vector_type(16))) __bf16 v16bf;
typedef __attribute__((ext_vector_type(8)))  __bf16 v8bf;
typedef __attribute__((ext_vector_type(8)))  float  v8f;

// ---------------------------------------------------------------------------
// Generic fill
// ---------------------------------------------------------------------------
__global__ void fill_kernel(float* __restrict__ p, float v, long n) {
    long i = (long)blockIdx.x * blockDim.x + threadIdx.x;
    if (i < n) p[i] = v;
}

// out[n][d] = bias[d]
__global__ void init_bias_kernel(float* __restrict__ out, const float* __restrict__ bias) {
    int n = blockIdx.x, d = threadIdx.x;
    out[(long)n * DD + d] = bias[d];
}

// ---------------------------------------------------------------------------
// One-shot f32 -> bf16 operand conversion (8 elements / thread)
// ---------------------------------------------------------------------------
__global__ void f32_to_bf16_kernel(const float* __restrict__ in, __bf16* __restrict__ out, long n) {
    long i = ((long)blockIdx.x * blockDim.x + threadIdx.x) * 8;
    if (i >= n) return;
    float4 a = *(const float4*)(in + i);
    float4 b = *(const float4*)(in + i + 4);
    v8bf o;
    o[0] = (__bf16)a.x; o[1] = (__bf16)a.y; o[2] = (__bf16)a.z; o[3] = (__bf16)a.w;
    o[4] = (__bf16)b.x; o[5] = (__bf16)b.y; o[6] = (__bf16)b.z; o[7] = (__bf16)b.w;
    *(v8bf*)(out + i) = o;
}

// ---------------------------------------------------------------------------
// x[n][d] = pre_gat_W[d][ids[n]] + b[d]   (pre_gat_W is [D, IN_SIZE] row-major)
// ---------------------------------------------------------------------------
__global__ void gather_x_kernel(const float* __restrict__ Wt, const float* __restrict__ b,
                                const int* __restrict__ ids, float* __restrict__ x) {
    int n = blockIdx.x, d = threadIdx.x;
    int id = ids[n];
    x[(long)n * DD + d] = Wt[(long)d * IN_SIZE + id] + b[d];
}

// ---------------------------------------------------------------------------
// WMMA GEMM (bf16 operands, f32 accumulate):
//   C[Mr,Nc] = A[Mr,K] * W[Nc,K]^T (+ bias[Nc])
// Grid = (Mr/128) x (Nc/64) blocks; block = 8 waves sharing one 64-col W panel.
// The panel (64 x K bf16, 32-64KB) is DMA'd into LDS with
// global_load_async_to_lds_b128 (ASYNCcnt) once per block; all 8 waves then
// read B fragments from LDS (ds_load_b128) while streaming A from L2.
// Wave w computes rows (mb*8+w)*16 .. +15 across the 64-col panel:
// 4 f32 accumulators, 4 WMMAs per A-fragment.
// Fragment layout per CDNA5 ISA 16-bit A(16x32)/B(32x16) maps:
//   lane L = (l=L&15, hi=L>>4); element j covers
//   k = k0 + (j<8 ? hi*8 + j : 16 + hi*8 + (j-8))  -> two 16B loads.
// ---------------------------------------------------------------------------
__device__ inline v16bf load_frag(const __bf16* __restrict__ p, int hi) {
    v8bf lo = *(const v8bf*)(p + hi * 8);
    v8bf hh = *(const v8bf*)(p + 16 + hi * 8);
    v16bf r;
#pragma unroll
    for (int j = 0; j < 8; ++j) { r[j] = lo[j]; r[8 + j] = hh[j]; }
    return r;
}

__device__ inline v16bf lds_frag(const __bf16* p, int hi) {
    v8bf lo = *(const v8bf*)(p + hi * 8);
    v8bf hh = *(const v8bf*)(p + 16 + hi * 8);
    v16bf r;
#pragma unroll
    for (int j = 0; j < 8; ++j) { r[j] = lo[j]; r[8 + j] = hh[j]; }
    return r;
}

__global__ void wmma_gemm_bf16_kernel(const __bf16* __restrict__ A, const __bf16* __restrict__ W,
                                      const float* __restrict__ bias, float* __restrict__ C,
                                      int Mr, int Nc, int K) {
    extern __shared__ __bf16 wpanel[];               // [64][K] bf16

    const int wave = threadIdx.x >> 5;
    const int lane = threadIdx.x & 31;
    const int l  = lane & 15;
    const int hi = lane >> 4;

    const int nPanels = Nc >> 6;
    const int mb = blockIdx.x / nPanels;             // 128-row block
    const int nt = blockIdx.x % nPanels;             // 64-col panel
    const long mt = (long)mb * 8 + wave;             // this wave's 16-row tile

    // ---- stage W panel (64 rows x K) into LDS via async DMA ----
    {
        const __bf16* __restrict__ gpanel = W + (long)nt * 64 * K;
        const int chunks = (64 * K) >> 3;            // 16B chunks
        for (int c = threadIdx.x; c < chunks; c += 256) {
            const __bf16* g = gpanel + c * 8;
            unsigned lds_off = (unsigned)(size_t)(&wpanel[c * 8]);
            asm volatile("global_load_async_to_lds_b128 %0, %1, off"
                         :: "v"(lds_off), "v"(g) : "memory");
        }
        asm volatile("s_wait_asynccnt 0" ::: "memory");
        __syncthreads();
    }

    const __bf16* __restrict__ Arow = A + (mt * 16 + l) * (long)K;
    const __bf16* Bl0 = wpanel + ( 0 + l) * K;
    const __bf16* Bl1 = wpanel + (16 + l) * K;
    const __bf16* Bl2 = wpanel + (32 + l) * K;
    const __bf16* Bl3 = wpanel + (48 + l) * K;

    v8f c0 = {}, c1 = {}, c2 = {}, c3 = {};
    for (int k0 = 0; k0 < K; k0 += 32) {
        if (k0 + 32 < K)
            __builtin_prefetch(Arow + k0 + 32, 0, 3);    // near-cache prefetch of next A step
        v16bf a  = load_frag(Arow + k0, hi);
        v16bf b0 = lds_frag(Bl0 + k0, hi);
        v16bf b1 = lds_frag(Bl1 + k0, hi);
        v16bf b2 = lds_frag(Bl2 + k0, hi);
        v16bf b3 = lds_frag(Bl3 + k0, hi);
        c0 = __builtin_amdgcn_wmma_f32_16x16x32_bf16(false, a, false, b0, (short)0, c0, false, false);
        c1 = __builtin_amdgcn_wmma_f32_16x16x32_bf16(false, a, false, b1, (short)0, c1, false, false);
        c2 = __builtin_amdgcn_wmma_f32_16x16x32_bf16(false, a, false, b2, (short)0, c2, false, false);
        c3 = __builtin_amdgcn_wmma_f32_16x16x32_bf16(false, a, false, b3, (short)0, c3, false, false);
    }

    // Epilogue: lane l owns column (nt*64 + nn*16 + l); VGPR r owns row mt*16 + hi*8 + r
#pragma unroll
    for (int nn = 0; nn < 4; ++nn) {
        const v8f* cp = nn == 0 ? &c0 : nn == 1 ? &c1 : nn == 2 ? &c2 : &c3;
        long col = (long)nt * 64 + nn * 16 + l;
        float bv = bias ? bias[col] : 0.0f;
#pragma unroll
        for (int r = 0; r < 8; ++r) {
            long row = mt * 16 + hi * 8 + r;
            C[row * (long)Nc + col] = (*cp)[r] + bv;
        }
    }
}

// ---------------------------------------------------------------------------
// a[n][h] = sum_c xl[n][h*64+c] * att[h][c]
// ---------------------------------------------------------------------------
__global__ void att_reduce_kernel(const float* __restrict__ xl, const float* __restrict__ att,
                                  float* __restrict__ out) {
    __shared__ float sm[256];
    const int n = blockIdx.x;
    const int t = threadIdx.x;
    sm[t] = xl[(long)n * DD + t] * att[t];
    __syncthreads();
#pragma unroll
    for (int s = 32; s >= 1; s >>= 1) {
        if ((t & 63) < s) sm[t] += sm[t + s];
        __syncthreads();
    }
    if ((t & 63) == 0) out[(long)n * HH + (t >> 6)] = sm[t];
}

// ---------------------------------------------------------------------------
// Edge softmax (scatter form, atomics into L2-resident segment buffers)
// ---------------------------------------------------------------------------
__device__ inline void atomicMaxF(float* addr, float value) {
    float old = __int_as_float(__hip_atomic_load((int*)addr, __ATOMIC_RELAXED,
                                                 __HIP_MEMORY_SCOPE_AGENT));
    while (old < value) {
        int assumed = __float_as_int(old);
        int prev = atomicCAS((int*)addr, assumed, __float_as_int(value));
        if (prev == assumed) break;
        old = __int_as_float(prev);
    }
}

__global__ void edge_logits_kernel(const float* __restrict__ as_, const float* __restrict__ ad_,
                                   const int* __restrict__ src, const int* __restrict__ dst,
                                   float* __restrict__ ebuf, float* __restrict__ emax, long E4) {
    long id = (long)blockIdx.x * 256 + threadIdx.x;
    if (id >= E4) return;
    int e = (int)(id >> 2), h = (int)(id & 3);
    float v = as_[(long)src[e] * HH + h] + ad_[(long)dst[e] * HH + h];
    v = v > 0.0f ? v : NEG_SLOPE * v;
    ebuf[id] = v;
    atomicMaxF(&emax[(long)dst[e] * HH + h], v);
}

__global__ void edge_exp_kernel(float* __restrict__ ebuf, const float* __restrict__ emax,
                                const int* __restrict__ dst, float* __restrict__ den, long E4) {
    long id = (long)blockIdx.x * 256 + threadIdx.x;
    if (id >= E4) return;
    int e = (int)(id >> 2), h = (int)(id & 3);
    float m = emax[(long)dst[e] * HH + h];
    if (m < -1e29f) m = 0.0f;
    float ee = __expf(ebuf[id] - m);
    ebuf[id] = ee;
    atomicAdd(&den[(long)dst[e] * HH + h], ee);
}

// out[dst] += xl[src] * (ee/(den+eps) * ew)  -- 64 threads/edge, 4 channels each
__global__ void scatter_msg_kernel(const float* __restrict__ xl, const float* __restrict__ ebuf,
                                   const float* __restrict__ den, const int* __restrict__ src,
                                   const int* __restrict__ dst, const float* __restrict__ ew,
                                   float* __restrict__ out) {
    const int e = blockIdx.x;
    const int t = threadIdx.x;
    const int s = src[e], d = dst[e];
    const int h = t >> 4;
    const float alpha = ebuf[(long)e * HH + h] /
                        (den[(long)d * HH + h] + 1e-16f) * ew[e];
    const float4 xv = *(const float4*)(xl + (long)s * DD + t * 4);
    float* op = out + (long)d * DD + t * 4;
    atomicAdd(op + 0, xv.x * alpha);
    atomicAdd(op + 1, xv.y * alpha);
    atomicAdd(op + 2, xv.z * alpha);
    atomicAdd(op + 3, xv.w * alpha);
}

// ---------------------------------------------------------------------------
// acc[b][d] += scale_i * interp_mask[b][i] * (2*x1[b][d] + 2*x2[b][d])
// ---------------------------------------------------------------------------
__global__ void accumulate_kernel(const float* __restrict__ x1, const float* __restrict__ x2,
                                  const float* __restrict__ masks, const float* __restrict__ iw,
                                  int mi, float* __restrict__ acc) {
    const int b = blockIdx.x, d = threadIdx.x;
    float w0 = iw[0], w1 = iw[1], w2 = iw[2];
    float mx = fmaxf(w0, fmaxf(w1, w2));
    float e0 = __expf(w0 - mx), e1 = __expf(w1 - mx), e2 = __expf(w2 - mx);
    float scale = (mi == 0 ? e0 : mi == 1 ? e1 : e2) / (e0 + e1 + e2);
    float m0 = masks[(long)b * MM + 0], m1 = masks[(long)b * MM + 1], m2 = masks[(long)b * MM + 2];
    float l0 = m0 + (1.0f - m0) * -1e10f;
    float l1 = m1 + (1.0f - m1) * -1e10f;
    float l2 = m2 + (1.0f - m2) * -1e10f;
    mx = fmaxf(l0, fmaxf(l1, l2));
    e0 = __expf(l0 - mx); e1 = __expf(l1 - mx); e2 = __expf(l2 - mx);
    float im = (mi == 0 ? e0 : mi == 1 ? e1 : e2) / (e0 + e1 + e2);

    long idx = (long)b * DD + d;
    float v = 2.0f * x1[idx] + 2.0f * x2[idx];
    acc[idx] += scale * im * v;
}

__global__ void write_scales_kernel(const float* __restrict__ iw, float* __restrict__ out) {
    float w0 = iw[0], w1 = iw[1], w2 = iw[2];
    float mx = fmaxf(w0, fmaxf(w1, w2));
    float e0 = __expf(w0 - mx), e1 = __expf(w1 - mx), e2 = __expf(w2 - mx);
    float s = e0 + e1 + e2;
    out[0] = e0 / s; out[1] = e1 / s; out[2] = e2 / s;
}

// ---------------------------------------------------------------------------
// Host launcher
// ---------------------------------------------------------------------------
static inline int gemm_blocks(int Mr, int Nc) {
    return (Mr / 128) * (Nc / 64);                   // 128x64 per block (8 waves)
}
static inline int cvt_blocks(long n) { return (int)((n / 8 + 255) / 256); }

extern "C" void kernel_launch(void* const* d_in, const int* in_sizes, int n_in,
                              void* d_out, int out_size, void* d_ws, size_t ws_size,
                              hipStream_t stream) {
    const float* masks   = (const float*)d_in[0];
    const int*   n_ids   = (const int*)  d_in[1];
    const int*   src0    = (const int*)  d_in[2];
    const int*   dst0    = (const int*)  d_in[3];
    const float* w0      = (const float*)d_in[4];
    const int*   src1    = (const int*)  d_in[5];
    const int*   dst1    = (const int*)  d_in[6];
    const float* w1      = (const float*)d_in[7];
    const float* preW    = (const float*)d_in[8];
    const float* preB    = (const float*)d_in[9];
    const float* linW    = (const float*)d_in[10];
    const float* attS    = (const float*)d_in[11];
    const float* attD    = (const float*)d_in[12];
    const float* gatB    = (const float*)d_in[13];
    const float* iw      = (const float*)d_in[14];
    const float* embW    = (const float*)d_in[15];
    const float* embB    = (const float*)d_in[16];

    float* out = (float*)d_out;
    const long OFF_EMB    = (long)NB * NB;
    const long OFF_SCALES = OFF_EMB + (long)NB * EMB;

    // workspace layout (f32 region, then bf16 region)
    float* w = (float*)d_ws;
    float* x    = w;
    float* xl   = x    + (long)N0 * DD;
    float* a_s  = xl   + (long)N0 * DD;
    float* a_d  = a_s  + (long)N0 * HH;
    float* ebuf = a_d  + (long)N1 * HH;
    float* emax = ebuf + (long)E0 * HH;
    float* den  = emax + (long)N1 * HH;
    float* x1   = den  + (long)N1 * HH;
    float* x2   = x1   + (long)N1 * DD;
    float* acc  = x2   + (long)NB * DD;
    __bf16* abf = (__bf16*)(acc + (long)NB * DD);   // [N0*256] bf16 (A operands)
    __bf16* wbf = abf + (long)N0 * DD;              // [NB*EMB] bf16 (B operands / emb)

    {
        long n = (long)NB * DD;
        fill_kernel<<<(int)((n + 255) / 256), 256, 0, stream>>>(acc, 0.0f, n);
    }

    for (int i = 0; i < MM; ++i) {
        const float* linWi = linW + (long)i * DD * DD;
        const float* attSi = attS + (long)i * DD;
        const float* attDi = attD + (long)i * DD;
        const float* gatBi = gatB + (long)i * DD;

        gather_x_kernel<<<N0, DD, 0, stream>>>(preW + (long)i * DD * IN_SIZE,
                                               preB + (long)i * DD,
                                               n_ids + (long)i * N0, x);

        // ---- hop 0: x[N0] -> x1[N1] ----
        f32_to_bf16_kernel<<<cvt_blocks((long)N0 * DD), 256, 0, stream>>>(x, abf, (long)N0 * DD);
        f32_to_bf16_kernel<<<cvt_blocks((long)DD * DD), 256, 0, stream>>>(linWi, wbf, (long)DD * DD);
        wmma_gemm_bf16_kernel<<<gemm_blocks(N0, DD), 256, 64 * DD * 2, stream>>>(
            abf, wbf, nullptr, xl, N0, DD, DD);
        att_reduce_kernel<<<N0, 256, 0, stream>>>(xl, attSi, a_s);
        att_reduce_kernel<<<N1, 256, 0, stream>>>(xl, attDi, a_d);
        fill_kernel<<<(N1 * HH + 255) / 256, 256, 0, stream>>>(emax, -1e30f, (long)N1 * HH);
        fill_kernel<<<(N1 * HH + 255) / 256, 256, 0, stream>>>(den, 0.0f, (long)N1 * HH);
        init_bias_kernel<<<N1, DD, 0, stream>>>(x1, gatBi);
        {
            long e4 = (long)E0 * HH;
            int blks = (int)((e4 + 255) / 256);
            edge_logits_kernel<<<blks, 256, 0, stream>>>(a_s, a_d, src0 + (long)i * E0,
                                                         dst0 + (long)i * E0, ebuf, emax, e4);
            edge_exp_kernel<<<blks, 256, 0, stream>>>(ebuf, emax, dst0 + (long)i * E0, den, e4);
            scatter_msg_kernel<<<E0, 64, 0, stream>>>(xl, ebuf, den, src0 + (long)i * E0,
                                                      dst0 + (long)i * E0, w0 + (long)i * E0, x1);
        }

        // ---- hop 1: x1[N1] -> x2[NB] ----
        f32_to_bf16_kernel<<<cvt_blocks((long)N1 * DD), 256, 0, stream>>>(x1, abf, (long)N1 * DD);
        wmma_gemm_bf16_kernel<<<gemm_blocks(N1, DD), 256, 64 * DD * 2, stream>>>(
            abf, wbf, nullptr, xl, N1, DD, DD);
        att_reduce_kernel<<<N1, 256, 0, stream>>>(xl, attSi, a_s);
        att_reduce_kernel<<<NB, 256, 0, stream>>>(xl, attDi, a_d);
        fill_kernel<<<(NB * HH + 255) / 256, 256, 0, stream>>>(emax, -1e30f, (long)NB * HH);
        fill_kernel<<<(NB * HH + 255) / 256, 256, 0, stream>>>(den, 0.0f, (long)NB * HH);
        init_bias_kernel<<<NB, DD, 0, stream>>>(x2, gatBi);
        {
            long e4 = (long)E1 * HH;
            int blks = (int)((e4 + 255) / 256);
            edge_logits_kernel<<<blks, 256, 0, stream>>>(a_s, a_d, src1 + (long)i * E1,
                                                         dst1 + (long)i * E1, ebuf, emax, e4);
            edge_exp_kernel<<<blks, 256, 0, stream>>>(ebuf, emax, dst1 + (long)i * E1, den, e4);
            scatter_msg_kernel<<<E1, 64, 0, stream>>>(xl, ebuf, den, src1 + (long)i * E1,
                                                      dst1 + (long)i * E1, w1 + (long)i * E1, x2);
        }

        accumulate_kernel<<<NB, DD, 0, stream>>>(x1, x2, masks, iw, i, acc);
    }

    // emb = acc @ emb_W.T + emb_b
    f32_to_bf16_kernel<<<cvt_blocks((long)NB * DD), 256, 0, stream>>>(acc, abf, (long)NB * DD);
    f32_to_bf16_kernel<<<cvt_blocks((long)EMB * DD), 256, 0, stream>>>(embW, wbf, (long)EMB * DD);
    wmma_gemm_bf16_kernel<<<gemm_blocks(NB, EMB), 256, 64 * DD * 2, stream>>>(
        abf, wbf, embB, out + OFF_EMB, NB, EMB, DD);
    // dot = emb @ emb.T
    f32_to_bf16_kernel<<<cvt_blocks((long)NB * EMB), 256, 0, stream>>>(out + OFF_EMB, wbf,
                                                                       (long)NB * EMB);
    wmma_gemm_bf16_kernel<<<gemm_blocks(NB, NB), 256, 64 * EMB * 2, stream>>>(
        wbf, wbf, nullptr, out, NB, NB, EMB);
    // scales
    write_scales_kernel<<<1, 1, 0, stream>>>(iw, out + OFF_SCALES);

    (void)in_sizes; (void)n_in; (void)out_size; (void)ws_size;
}
